// PointTransformer_RoPE_10831907520674
// MI455X (gfx1250) — matrix operand order repaired and checked by reading, not compiled, not run on hardware
//
#include <hip/hip_runtime.h>
#include <math.h>

// ---------------------------------------------------------------------------
// Point Transformer w/ 3-axis RoPE — CDNA5 (gfx1250).
// All dense math via V_WMMA_F32_16X16X32_F16 (f16 operands, f32 accumulate).
// Weights kept as transposed, zero-padded f16 mirrors Wt[Kp][Np] so every B
// fragment k row spans 16 contiguous halfs across lanes (lane = N column), stride Np per lane; no selects.
// One wave = 16x64 output tile (4 accumulators).
// ---------------------------------------------------------------------------

typedef __attribute__((ext_vector_type(16))) _Float16 v16h;
typedef __attribute__((ext_vector_type(8)))  float    v8f;

#define Bb    32
#define NPTS  2048
#define Gg    128
#define Mm    32
#define Cc    384
#define DEPTH 12
#define Hh    8
#define HD    48
#define NTOK  129     // 1 CLS + 128 groups
#define MLPD  1536

// K index inside a 32-wide K step for the f16 A operand (ISA 16x32 A table).
__device__ __forceinline__ int wmma_ka(int lane, int e) {
  return ((e >> 3) << 4) + (((lane >> 4) & 1) << 3) + (e & 7);
}

// ---------------------------------------------------------------------------
// Transpose-convert: src (L, N, K) f32 row-major  ->  dst (L, Kp, Np) f16,
// dst[l][k][n] = src[l][n][k], zero-padded for k>=K or n>=N.
// ---------------------------------------------------------------------------
__global__ void cvt_t_kernel(const float* __restrict__ src, _Float16* __restrict__ dst,
                             int L, int N, int K, int Np, int Kp) {
  int i = blockIdx.x * 256 + threadIdx.x;
  int per = Kp * Np;
  if (i >= L * per) return;
  int l = i / per, r = i % per;
  int k = r / Np, n = r % Np;
  float v = (k < K && n < N) ? src[(size_t)l * N * K + (size_t)n * K + k] : 0.f;
  dst[i] = (_Float16)v;
}

// ---------------------------------------------------------------------------
// Generic GEMM: val = act( scale*( A[M,K]f16 @ W^T + bias ) + shift )
//               out(f32) = resid + val   and/or   out16(f16) = val
// Wt is the transposed padded mirror (Kp=K rows here, Np cols, Np%64==0).
// One wave = one 16x64 tile (4 accumulators). M%16==0, K%32==0 at all sites.
// act: 0=none 1=relu 2=gelu(exact)
// ---------------------------------------------------------------------------
__global__ void gemm_kernel(const _Float16* __restrict__ A, const _Float16* __restrict__ Wt,
                            const float* __restrict__ bias,
                            const float* __restrict__ scale, const float* __restrict__ shift,
                            const float* __restrict__ resid,
                            float* __restrict__ out, _Float16* __restrict__ out16,
                            int M, int N, int K, int Np, int act) {
  int lane = threadIdx.x & 31;
  int wave = threadIdx.x >> 5;
  int tn = Np >> 6;                       // 64-wide column groups
  int tm = M >> 4;
  int tile = blockIdx.x * 8 + wave;
  if (tile >= tm * tn) return;            // wave-uniform exit
  int row0 = (tile / tn) << 4;
  int col0 = (tile % tn) << 6;

  v8f acc[4] = {};
  const int m = row0 + (lane & 15);
  const _Float16* Arow = A + (size_t)m * K;
  for (int kb = 0; kb < K; kb += 32) {
    v16h a;
#pragma unroll
    for (int e = 0; e < 16; ++e) {        // two runs of 8 consecutive halfs
      a[e] = Arow[kb + wmma_ka(lane, e)];
    }
    __builtin_prefetch(Arow + kb + 64, 0, 1);   // global_prefetch_b8 (speculative)
    const _Float16* Wk = Wt + (size_t)kb * Np + col0 + (lane & 15);  // lane = N column
#pragma unroll
    for (int s = 0; s < 4; ++s) {
      v16h bB;
#pragma unroll
      for (int e = 0; e < 16; ++e) bB[e] = Wk[(size_t)wmma_ka(lane, e) * Np + s * 16];  // k in A table order
      acc[s] = __builtin_amdgcn_wmma_f32_16x16x32_f16(false, a, false, bB, (short)0,
                                                      acc[s], false, false);
    }
  }
  int hi = lane >> 4;
#pragma unroll
  for (int s = 0; s < 4; ++s) {
    int n = col0 + s * 16 + (lane & 15);
    if (n < N) {                          // divergence only after all WMMAs
      float bz = bias ? bias[n] : 0.f;
      float sc = scale ? scale[n] : 1.f;
      float sh = shift ? shift[n] : 0.f;
#pragma unroll
      for (int v = 0; v < 8; ++v) {
        int mr = row0 + v + hi * 8;
        float val = acc[s][v] + bz;
        val = val * sc + sh;
        if (act == 1) val = fmaxf(val, 0.f);
        else if (act == 2) val = 0.5f * val * (1.f + erff(val * 0.7071067811865475f));
        if (out) {
          float r = val;
          if (resid) r += resid[(size_t)mr * N + n];
          out[(size_t)mr * N + n] = r;
        }
        if (out16) out16[(size_t)mr * N + n] = (_Float16)val;
      }
    }
  }
}

// ---------------------------------------------------------------------------
// FPS: one block per batch. Sequential 128 argmax rounds, pts cached in LDS.
// ---------------------------------------------------------------------------
__global__ void fps_kernel(const float* __restrict__ pts, int* __restrict__ cidx,
                           float* __restrict__ center) {
  __shared__ float px[NPTS], py[NPTS], pz[NPTS], dist[NPTS];
  __shared__ float rmax[256];
  __shared__ int   ridx[256];
  __shared__ int   s_last;
  int b = blockIdx.x, tid = threadIdx.x;
  const float* P = pts + (size_t)b * NPTS * 3;
  for (int i = tid; i < NPTS; i += 256) {
    px[i] = P[i * 3 + 0]; py[i] = P[i * 3 + 1]; pz[i] = P[i * 3 + 2];
    dist[i] = 1e10f;
  }
  if (tid == 0) s_last = 0;
  __syncthreads();
  for (int it = 0; it < Gg; ++it) {
    int last = s_last;
    if (tid == 0) {
      cidx[b * Gg + it] = last;
      center[(b * Gg + it) * 3 + 0] = px[last];
      center[(b * Gg + it) * 3 + 1] = py[last];
      center[(b * Gg + it) * 3 + 2] = pz[last];
    }
    float lx = px[last], ly = py[last], lz = pz[last];
    float bmax = -1.f; int bidx = 0;
    for (int i = tid; i < NPTS; i += 256) {
      float dx = px[i] - lx, dy = py[i] - ly, dz = pz[i] - lz;
      float dd = fminf(dist[i], dx * dx + dy * dy + dz * dz);
      dist[i] = dd;
      if (dd > bmax) { bmax = dd; bidx = i; }
    }
    rmax[tid] = bmax; ridx[tid] = bidx;
    __syncthreads();
    for (int s = 128; s > 0; s >>= 1) {
      if (tid < s) {
        float om = rmax[tid + s]; int oi = ridx[tid + s];
        if (om > rmax[tid] || (om == rmax[tid] && oi < ridx[tid])) {
          rmax[tid] = om; ridx[tid] = oi;
        }
      }
      __syncthreads();
    }
    if (tid == 0) s_last = ridx[0];
    __syncthreads();
  }
}

// ---------------------------------------------------------------------------
// KNN: one block per (b,g). 32 iterative argmin extractions (top_k tie order),
// writes centered neighborhoods (BG, 32, 3).
// ---------------------------------------------------------------------------
__global__ void knn_kernel(const float* __restrict__ pts, const float* __restrict__ center,
                           float* __restrict__ neigh) {
  __shared__ float d2[NPTS];
  __shared__ float rmin[256];
  __shared__ int   ridx[256];
  int bg = blockIdx.x;
  int b = bg >> 7;
  int tid = threadIdx.x;
  const float* P = pts + (size_t)b * NPTS * 3;
  float cx = center[bg * 3 + 0], cy = center[bg * 3 + 1], cz = center[bg * 3 + 2];
  for (int i = tid; i < NPTS; i += 256) {
    float dx = P[i * 3 + 0] - cx, dy = P[i * 3 + 1] - cy, dz = P[i * 3 + 2] - cz;
    d2[i] = dx * dx + dy * dy + dz * dz;
  }
  __syncthreads();
  for (int m = 0; m < Mm; ++m) {
    float bmin = 1e30f; int bidx = 0;
    for (int i = tid; i < NPTS; i += 256) {
      float d = d2[i];
      if (d < bmin) { bmin = d; bidx = i; }
    }
    rmin[tid] = bmin; ridx[tid] = bidx;
    __syncthreads();
    for (int s = 128; s > 0; s >>= 1) {
      if (tid < s) {
        float om = rmin[tid + s]; int oi = ridx[tid + s];
        if (om < rmin[tid] || (om == rmin[tid] && oi < ridx[tid])) {
          rmin[tid] = om; ridx[tid] = oi;
        }
      }
      __syncthreads();
    }
    if (tid == 0) {
      int j = ridx[0];
      neigh[((size_t)bg * Mm + m) * 3 + 0] = P[j * 3 + 0] - cx;
      neigh[((size_t)bg * Mm + m) * 3 + 1] = P[j * 3 + 1] - cy;
      neigh[((size_t)bg * Mm + m) * 3 + 2] = P[j * 3 + 2] - cz;
      d2[j] = 1e30f;
    }
    __syncthreads();
  }
}

// ---------------------------------------------------------------------------
// Encoder block GEMM: Os[32,N](f16) = act(affine(As[32,Kp] @ Wt + bias))
// A in LDS f16 (zero-padded); Wt global transposed padded f16 (Kp x Np=N).
// All operand loads unconditional.
// ---------------------------------------------------------------------------
__device__ __forceinline__ void blk_gemm(const _Float16* As, int lda, int Kp,
                                         const _Float16* Wt, const float* bias,
                                         const float* scale, const float* shift, int relu,
                                         _Float16* Os, int N) {
  int lane = threadIdx.x & 31;
  int wave = threadIdx.x >> 5;
  int tiles = 2 * (N >> 4);
  int kiters = Kp >> 5;
  for (int t = wave; t < tiles; t += 8) {
    int row0 = (t & 1) << 4;
    int col0 = (t >> 1) << 4;
    v8f acc = {};
    for (int ki = 0; ki < kiters; ++ki) {
      int kb = ki << 5;
      v16h a, bB;
      int m = row0 + (lane & 15);
#pragma unroll
      for (int e = 0; e < 16; ++e) a[e] = As[m * lda + kb + wmma_ka(lane, e)];
      const _Float16* Wk = Wt + (size_t)kb * N + col0 + (lane & 15);
#pragma unroll
      for (int e = 0; e < 16; ++e) bB[e] = Wk[(size_t)wmma_ka(lane, e) * N];   // k in A table order
      acc = __builtin_amdgcn_wmma_f32_16x16x32_f16(false, a, false, bB, (short)0, acc,
                                                   false, false);
    }
    int hi = lane >> 4;
    int n = col0 + (lane & 15);
#pragma unroll
    for (int v = 0; v < 8; ++v) {
      int m = row0 + v + hi * 8;
      float val = acc[v] + (bias ? bias[n] : 0.f);
      if (scale) val = val * scale[n] + shift[n];
      if (relu) val = fmaxf(val, 0.f);
      Os[m * N + n] = (_Float16)val;
    }
  }
}

// ---------------------------------------------------------------------------
// Fused mini-PointNet encoder: one block per group, all stages in LDS (2x32KB).
// ---------------------------------------------------------------------------
__global__ void enc_kernel(const float* __restrict__ neigh,
                           const _Float16* W1, const float* b1, const float* g1, const float* be1,
                           const _Float16* W2, const float* b2,
                           const _Float16* W3, const float* b3, const float* g2, const float* be2,
                           const _Float16* W4, const float* b4,
                           float* __restrict__ tok) {
  extern __shared__ char smem[];
  _Float16* buf0 = (_Float16*)smem;              // 16384 halves (32 KB)
  _Float16* buf1 = (_Float16*)(smem + 32768);    // 16384 halves (32 KB)
  int bg = blockIdx.x, tid = threadIdx.x;

  // neigh -> buf0 (32 rows x lda 32, zero padded)
  for (int i = tid; i < 32 * 32; i += 256) buf0[i] = (_Float16)0.f;
  __syncthreads();
  for (int i = tid; i < 32 * 3; i += 256) {
    int m = i / 3, j = i % 3;
    buf0[m * 32 + j] = (_Float16)neigh[(size_t)bg * 96 + i];
  }
  __syncthreads();
  blk_gemm(buf0, 32, 32, W1, b1, g1, be1, 1, buf1, 128);              // f1 (32x128)
  __syncthreads();
  blk_gemm(buf1, 128, 128, W2, b2, nullptr, nullptr, 0, buf0, 256);   // f2 (32x256)
  __syncthreads();
  // cat = [broadcast(max_m f2) | f2]  -> buf1 (32x512)
  for (int c = tid; c < 256; c += 256) {
    float mx = -1e30f;
    for (int m = 0; m < 32; ++m) mx = fmaxf(mx, (float)buf0[m * 256 + c]);
    for (int m = 0; m < 32; ++m) {
      buf1[m * 512 + c]       = (_Float16)mx;
      buf1[m * 512 + 256 + c] = buf0[m * 256 + c];
    }
  }
  __syncthreads();
  blk_gemm(buf1, 512, 512, W3, b3, g2, be2, 1, buf0, 512);            // f3 (32x512)
  __syncthreads();
  blk_gemm(buf0, 512, 512, W4, b4, nullptr, nullptr, 0, buf1, 384);   // f4 (32x384)
  __syncthreads();
  for (int c = tid; c < 384; c += 256) {
    float mx = -1e30f;
    for (int m = 0; m < 32; ++m) mx = fmaxf(mx, (float)buf1[m * 384 + c]);
    tok[(size_t)bg * 384 + c] = mx;
  }
}

// Axial RoPE angle tables: pcos/psin (B, G, 24), layout axis*8 + j.
__global__ void ang_kernel(const float* __restrict__ center, float* __restrict__ pcos,
                           float* __restrict__ psin) {
  int idx = blockIdx.x * 256 + threadIdx.x;
  if (idx >= Bb * Gg * 24) return;
  int p = idx % 24;
  int axis = p >> 3, j = p & 7;
  int bg = idx / 24;
  float fr = powf(100.f, -(float)j * 0.125f);
  float a = center[bg * 3 + axis] * fr;
  pcos[idx] = cosf(a);
  psin[idx] = sinf(a);
}

// x0 = [cls_token+cls_pos | tokens]
__global__ void x0_kernel(const float* __restrict__ tok, const float* __restrict__ cls_t,
                          const float* __restrict__ cls_p, float* __restrict__ x) {
  int idx = blockIdx.x * 256 + threadIdx.x;
  if (idx >= Bb * NTOK * Cc) return;
  int c = idx % Cc;
  int t = (idx / Cc) % NTOK;
  int b = idx / (Cc * NTOK);
  x[idx] = (t == 0) ? (cls_t[c] + cls_p[c])
                    : tok[((size_t)(b * Gg + t - 1)) * Cc + c];
}

// ---------------------------------------------------------------------------
// LayerNorm over C=384, one block (128 threads) per row; emits f16 (GEMM input).
// ---------------------------------------------------------------------------
__global__ void ln16_kernel(const float* __restrict__ x, const float* __restrict__ g,
                            const float* __restrict__ b, _Float16* __restrict__ out) {
  __shared__ float s1[128], s2[128];
  int row = blockIdx.x, tid = threadIdx.x;
  const float* xr = x + (size_t)row * Cc;
  float a0 = xr[tid], a1 = xr[tid + 128], a2 = xr[tid + 256];
  s1[tid] = a0 + a1 + a2;
  s2[tid] = a0 * a0 + a1 * a1 + a2 * a2;
  __syncthreads();
  for (int s = 64; s > 0; s >>= 1) {
    if (tid < s) { s1[tid] += s1[tid + s]; s2[tid] += s2[tid + s]; }
    __syncthreads();
  }
  float mean = s1[0] * (1.f / 384.f);
  float var  = s2[0] * (1.f / 384.f) - mean * mean;
  float inv = rsqrtf(var + 1e-5f);
  _Float16* o = out + (size_t)row * Cc;
  o[tid]       = (_Float16)((a0 - mean) * inv * g[tid]       + b[tid]);
  o[tid + 128] = (_Float16)((a1 - mean) * inv * g[tid + 128] + b[tid + 128]);
  o[tid + 256] = (_Float16)((a2 - mean) * inv * g[tid + 256] + b[tid + 256]);
}

// ---------------------------------------------------------------------------
// Givens rotation (all tokens) + positional RoPE (non-CLS) on q and k in the
// packed qkv buffer (B*129, 1152), f32 in place.
// ---------------------------------------------------------------------------
__global__ void rope_kernel(float* __restrict__ qkv, const float* __restrict__ theta,
                            const float* __restrict__ pcos, const float* __restrict__ psin) {
  int idx = blockIdx.x * 256 + threadIdx.x;
  if (idx >= Bb * NTOK * Hh * 24) return;
  int p  = idx % 24;
  int hh = (idx / 24) % Hh;
  int t  = (idx / (24 * Hh)) % NTOK;
  int b  = idx / (24 * Hh * NTOK);
  float th = theta[hh * 24 + p];
  float gc = cosf(th), gs = sinf(th);
  int pi = (b * Gg + ((t > 0) ? (t - 1) : 0)) * 24 + p;
  float pc = (t > 0) ? pcos[pi] : 1.f;
  float ps = (t > 0) ? psin[pi] : 0.f;
  size_t row = (size_t)(b * NTOK + t) * (3 * Cc);
#pragma unroll
  for (int s = 0; s < 2; ++s) {          // q (s=0) and k (s=1)
    size_t off = row + (size_t)s * Cc + hh * HD + 2 * p;
    float xr = qkv[off], xi = qkv[off + 1];
    float r1 = xr * gc - xi * gs, i1 = xr * gs + xi * gc;
    qkv[off]     = r1 * pc - i1 * ps;
    qkv[off + 1] = r1 * ps + i1 * pc;
  }
}

// ---------------------------------------------------------------------------
// Attention: one block per (b, head, 16-row tile). S=QK^T (WMMA), softmax,
// O=P.V (WMMA). K-matrix stored transposed in LDS; B fragment elements at stride 144 (lane = token).
// All operand reads unconditional. grid = 2304 blocks, 256 thr.
// ---------------------------------------------------------------------------
__global__ void attn_kernel(const float* __restrict__ qkv, _Float16* __restrict__ o16) {
  __shared__ _Float16 sq[16 * 64];     // (row, k) K padded 48->64
  __shared__ _Float16 skT[64 * 144];   // (k, token): transposed K-matrix
  __shared__ _Float16 sv[160 * HD];    // (token, d), rows padded 129->160
  __shared__ float    S[16 * 160];     // cols padded 144->160
  int blk = blockIdx.x;
  int rt = blk % 9;
  int h  = (blk / 9) % Hh;
  int b  = blk / (9 * Hh);
  int tid = threadIdx.x;
  const float scale = 0.14433756729740643f;   // 48^-0.5

  for (int i = tid; i < 64 * 144; i += 256) {
    int k = i / 144, t = i % 144;
    int kk = (k < HD) ? k : 0;
    int tt = (t < NTOK) ? t : 0;
    float kv = qkv[(size_t)(b * NTOK + tt) * (3 * Cc) + Cc + h * HD + kk];
    skT[i] = (_Float16)((k < HD && t < NTOK) ? kv : 0.f);
  }
  for (int i = tid; i < 160 * HD; i += 256) {
    int t = i / HD, d = i % HD;
    int tt = (t < NTOK) ? t : 0;
    float vv = qkv[(size_t)(b * NTOK + tt) * (3 * Cc) + 2 * Cc + h * HD + d];
    sv[i] = (_Float16)((t < NTOK) ? vv : 0.f);
  }
  for (int i = tid; i < 16 * 64; i += 256) {
    int r = i >> 6, d = i & 63;
    int t = rt * 16 + r;
    int tt = (t < NTOK) ? t : 0;
    int dd = (d < HD) ? d : 0;
    float qv = qkv[(size_t)(b * NTOK + tt) * (3 * Cc) + h * HD + dd];
    sq[i] = (_Float16)((t < NTOK && d < HD) ? qv : 0.f);
  }
  __syncthreads();

  // S[16,144] = q @ k^T * scale   (K = 64 padded, zero tail)
  {
    int lane = tid & 31, wave = tid >> 5;
    for (int ct = wave; ct < 9; ct += 8) {
      int col0 = ct * 16;
      v8f acc = {};
      for (int kb = 0; kb < 64; kb += 32) {
        v16h a, bB;
        int m = lane & 15;
#pragma unroll
        for (int e = 0; e < 16; ++e) a[e] = sq[m * 64 + kb + wmma_ka(lane, e)];
        const _Float16* Kk = skT + kb * 144 + col0 + (lane & 15);
#pragma unroll
        for (int e = 0; e < 16; ++e) bB[e] = Kk[wmma_ka(lane, e) * 144];   // k in A table order
        acc = __builtin_amdgcn_wmma_f32_16x16x32_f16(false, a, false, bB, (short)0, acc,
                                                     false, false);
      }
      int hi = lane >> 4, n = col0 + (lane & 15);
#pragma unroll
      for (int v = 0; v < 8; ++v) S[(v + hi * 8) * 160 + n] = acc[v] * scale;
    }
  }
  __syncthreads();

  // softmax per row (valid keys 0..128); zero padded cols so P@V tail adds 0.
  if (tid < 16) {
    int t = rt * 16 + tid;
    float* row = S + tid * 160;
    if (t < NTOK) {
      float mx = -1e30f;
      for (int j = 0; j < NTOK; ++j) mx = fmaxf(mx, row[j]);
      float s = 0.f;
      for (int j = 0; j < NTOK; ++j) { float e = expf(row[j] - mx); row[j] = e; s += e; }
      float inv = 1.f / s;
      for (int j = 0; j < NTOK; ++j) row[j] *= inv;
      for (int j = NTOK; j < 160; ++j) row[j] = 0.f;
    } else {
      for (int j = 0; j < 160; ++j) row[j] = 0.f;
    }
  }
  __syncthreads();

  // O[16,48] = P @ V   (K = 160 padded, zero tail), store f16 for proj GEMM.
  {
    int lane = tid & 31, wave = tid >> 5;
    for (int ct = wave; ct < 3; ct += 8) {
      int col0 = ct * 16;
      v8f acc = {};
      for (int kb = 0; kb < 160; kb += 32) {
        v16h a, bB;
        int m = lane & 15;
#pragma unroll
        for (int e = 0; e < 16; ++e)
          a[e] = (_Float16)S[m * 160 + kb + wmma_ka(lane, e)];
        const _Float16* Vk = sv + kb * HD + col0 + (lane & 15);
#pragma unroll
        for (int e = 0; e < 16; ++e) bB[e] = Vk[wmma_ka(lane, e) * HD];   // k in A table order
        acc = __builtin_amdgcn_wmma_f32_16x16x32_f16(false, a, false, bB, (short)0, acc,
                                                     false, false);
      }
      int hi = lane >> 4, n = col0 + (lane & 15);
#pragma unroll
      for (int v = 0; v < 8; ++v) {
        int m = v + hi * 8;
        int t = rt * 16 + m;
        if (t < NTOK)
          o16[(size_t)(b * NTOK + t) * Cc + h * HD + n] = (_Float16)acc[v];
      }
    }
  }
}

// feat16 = [ h[:,0] | max over tokens 1..128 of h ]  (B, 768) f16
__global__ void feat_kernel(const _Float16* __restrict__ h, _Float16* __restrict__ feat) {
  int idx = blockIdx.x * 256 + threadIdx.x;
  if (idx >= Bb * 2 * Cc) return;
  int b = idx / (2 * Cc), c = idx % (2 * Cc);
  const _Float16* xb = h + (size_t)b * NTOK * Cc;
  if (c < Cc) {
    feat[idx] = xb[c];
  } else {
    int cc = c - Cc;
    float mx = -1e30f;
    for (int t = 1; t < NTOK; ++t) mx = fmaxf(mx, (float)xb[(size_t)t * Cc + cc]);
    feat[idx] = (_Float16)mx;
  }
}

// ---------------------------------------------------------------------------
extern "C" void kernel_launch(void* const* d_in, const int* in_sizes, int n_in,
                              void* d_out, int out_size, void* d_ws, size_t ws_size,
                              hipStream_t stream) {
  (void)in_sizes; (void)n_in; (void)out_size; (void)ws_size;
  const float* pts      = (const float*)d_in[0];
  const float* enc_w1   = (const float*)d_in[1];
  const float* enc_b1   = (const float*)d_in[2];
  const float* enc_g1   = (const float*)d_in[3];
  const float* enc_be1  = (const float*)d_in[4];
  const float* enc_w2   = (const float*)d_in[5];
  const float* enc_b2   = (const float*)d_in[6];
  const float* enc_w3   = (const float*)d_in[7];
  const float* enc_b3   = (const float*)d_in[8];
  const float* enc_g2   = (const float*)d_in[9];
  const float* enc_be2  = (const float*)d_in[10];
  const float* enc_w4   = (const float*)d_in[11];
  const float* enc_b4   = (const float*)d_in[12];
  const float* cls_tok  = (const float*)d_in[13];
  const float* cls_pos  = (const float*)d_in[14];
  const float* ln1_g    = (const float*)d_in[15];
  const float* ln1_b    = (const float*)d_in[16];
  const float* qkv_w    = (const float*)d_in[17];
  const float* gtheta   = (const float*)d_in[18];
  const float* proj_w   = (const float*)d_in[19];
  const float* proj_b   = (const float*)d_in[20];
  const float* ln2_g    = (const float*)d_in[21];
  const float* ln2_b    = (const float*)d_in[22];
  const float* fc1_w    = (const float*)d_in[23];
  const float* fc1_b    = (const float*)d_in[24];
  const float* fc2_w    = (const float*)d_in[25];
  const float* fc2_b    = (const float*)d_in[26];
  const float* norm_g   = (const float*)d_in[27];
  const float* norm_b   = (const float*)d_in[28];
  const float* hw1      = (const float*)d_in[29];
  const float* hb1      = (const float*)d_in[30];
  const float* hg1      = (const float*)d_in[31];
  const float* hbe1     = (const float*)d_in[32];
  const float* hw2      = (const float*)d_in[33];
  const float* hb2      = (const float*)d_in[34];
  const float* hg2      = (const float*)d_in[35];
  const float* hbe2     = (const float*)d_in[36];
  const float* hw3      = (const float*)d_in[37];
  const float* hb3      = (const float*)d_in[38];
  float* out = (float*)d_out;

  // ---- workspace carving ----
  char* wsb = (char*)d_ws;
  size_t off = 0;
  auto takeF = [&](size_t n) { float* p = (float*)(wsb + off); off += n * 4; return p; };
  auto takeH = [&](size_t n) {
    _Float16* p = (_Float16*)(wsb + off); off += ((n * 2 + 3) & ~(size_t)3); return p;
  };
  float*    neigh  = takeF((size_t)Bb * Gg * Mm * 3);
  float*    center = takeF((size_t)Bb * Gg * 3);
  int*      cidx   = (int*)takeF((size_t)Bb * Gg);
  float*    tok    = takeF((size_t)Bb * Gg * Cc);
  float*    x      = takeF((size_t)Bb * NTOK * Cc);
  float*    qkvb   = takeF((size_t)Bb * NTOK * 3 * Cc);
  float*    pc     = takeF((size_t)Bb * Gg * 24);
  float*    ps     = takeF((size_t)Bb * Gg * 24);
  _Float16* h16    = takeH((size_t)Bb * NTOK * Cc);
  _Float16* o16    = takeH((size_t)Bb * NTOK * Cc);
  _Float16* m1h    = takeH((size_t)Bb * NTOK * MLPD);
  _Float16* feat16 = takeH((size_t)Bb * 2 * Cc);
  _Float16* h1h    = takeH((size_t)Bb * 256);
  _Float16* h2h    = takeH((size_t)Bb * 256);
  // transposed, padded f16 weight mirrors: Wt[l][Kp][Np]
  _Float16* wq  = takeH((size_t)DEPTH * Cc * (3 * Cc));     // 384 x 1152
  _Float16* wp  = takeH((size_t)DEPTH * Cc * Cc);           // 384 x 384
  _Float16* w1m = takeH((size_t)DEPTH * Cc * MLPD);         // 384 x 1536
  _Float16* w2m = takeH((size_t)DEPTH * MLPD * Cc);         // 1536 x 384
  _Float16* we1 = takeH((size_t)32 * 128);                  // Kp 3->32
  _Float16* we2 = takeH((size_t)128 * 256);
  _Float16* we3 = takeH((size_t)512 * 512);
  _Float16* we4 = takeH((size_t)512 * Cc);
  _Float16* wh1 = takeH((size_t)(2 * Cc) * 256);
  _Float16* wh2 = takeH((size_t)256 * 256);
  _Float16* wh3 = takeH((size_t)256 * 64);                  // Np 40->64

  auto cvt_t = [&](const float* s, _Float16* d, int L, int N, int K, int Np, int Kp) {
    size_t tot = (size_t)L * Kp * Np;
    cvt_t_kernel<<<(int)((tot + 255) / 256), 256, 0, stream>>>(s, d, L, N, K, Np, Kp);
  };
  auto gemm = [&](const _Float16* A, const _Float16* Wt, const float* bias,
                  const float* scale, const float* shift, const float* resid,
                  float* o_, _Float16* o16_, int M, int N, int K, int Np, int act) {
    int tiles = (M / 16) * (Np / 64);
    gemm_kernel<<<(tiles + 7) / 8, 256, 0, stream>>>(A, Wt, bias, scale, shift, resid,
                                                     o_, o16_, M, N, K, Np, act);
  };

  // ---- weight transpose-conversions ----
  cvt_t(qkv_w, wq, DEPTH, 3 * Cc, Cc, 3 * Cc, Cc);
  cvt_t(proj_w, wp, DEPTH, Cc, Cc, Cc, Cc);
  cvt_t(fc1_w, w1m, DEPTH, MLPD, Cc, MLPD, Cc);
  cvt_t(fc2_w, w2m, DEPTH, Cc, MLPD, Cc, MLPD);
  cvt_t(enc_w1, we1, 1, 128, 3, 128, 32);
  cvt_t(enc_w2, we2, 1, 256, 128, 256, 128);
  cvt_t(enc_w3, we3, 1, 512, 512, 512, 512);
  cvt_t(enc_w4, we4, 1, Cc, 512, Cc, 512);
  cvt_t(hw1, wh1, 1, 256, 2 * Cc, 256, 2 * Cc);
  cvt_t(hw2, wh2, 1, 256, 256, 256, 256);
  cvt_t(hw3, wh3, 1, 40, 256, 64, 256);

  // ---- grouping ----
  fps_kernel<<<Bb, 256, 0, stream>>>(pts, cidx, center);
  knn_kernel<<<Bb * Gg, 256, 0, stream>>>(pts, center, neigh);

  // ---- mini-PointNet encoder (fused, 64KB dynamic LDS) ----
  (void)hipFuncSetAttribute((const void*)enc_kernel,
                            hipFuncAttributeMaxDynamicSharedMemorySize, 65536);
  enc_kernel<<<Bb * Gg, 256, 65536, stream>>>(neigh,
      we1, enc_b1, enc_g1, enc_be1, we2, enc_b2,
      we3, enc_b3, enc_g2, enc_be2, we4, enc_b4, tok);

  // ---- RoPE tables and token assembly ----
  ang_kernel<<<(Bb * Gg * 24 + 255) / 256, 256, 0, stream>>>(center, pc, ps);
  x0_kernel<<<(Bb * NTOK * Cc + 255) / 256, 256, 0, stream>>>(tok, cls_tok, cls_pos, x);

  const int ROWS = Bb * NTOK;  // 4128
  // ---- transformer layers ----
  for (int i = 0; i < DEPTH; ++i) {
    ln16_kernel<<<ROWS, 128, 0, stream>>>(x, ln1_g + i * Cc, ln1_b + i * Cc, h16);
    gemm(h16, wq + (size_t)i * Cc * 3 * Cc, nullptr, nullptr, nullptr, nullptr,
         qkvb, nullptr, ROWS, 3 * Cc, Cc, 3 * Cc, 0);
    rope_kernel<<<(Bb * NTOK * Hh * 24 + 255) / 256, 256, 0, stream>>>(
        qkvb, gtheta + i * Hh * 24, pc, ps);
    attn_kernel<<<Bb * Hh * 9, 256, 0, stream>>>(qkvb, o16);
    gemm(o16, wp + (size_t)i * Cc * Cc, proj_b + i * Cc, nullptr, nullptr,
         x, x, nullptr, ROWS, Cc, Cc, Cc, 0);
    ln16_kernel<<<ROWS, 128, 0, stream>>>(x, ln2_g + i * Cc, ln2_b + i * Cc, h16);
    gemm(h16, w1m + (size_t)i * Cc * MLPD, fc1_b + i * MLPD, nullptr, nullptr,
         nullptr, nullptr, m1h, ROWS, MLPD, Cc, MLPD, 2);
    gemm(m1h, w2m + (size_t)i * MLPD * Cc, fc2_b + i * Cc, nullptr, nullptr,
         x, x, nullptr, ROWS, Cc, MLPD, Cc, 0);
  }

  // ---- final norm + head ----
  ln16_kernel<<<ROWS, 128, 0, stream>>>(x, norm_g, norm_b, h16);
  feat_kernel<<<(Bb * 2 * Cc + 255) / 256, 256, 0, stream>>>(h16, feat16);
  gemm(feat16, wh1, hb1, hg1, hbe1, nullptr, nullptr, h1h, Bb, 256, 2 * Cc, 256, 1);
  gemm(h1h,    wh2, hb2, hg2, hbe2, nullptr, nullptr, h2h, Bb, 256, 256, 256, 1);
  gemm(h2h,    wh3, hb3, nullptr, nullptr, nullptr, out, nullptr, Bb, 40, 256, 64, 0);
}
